// _ChannelTopologyRefinement_89335319757567
// MI455X (gfx1250) — matrix operand order repaired and checked by reading, not compiled, not run on hardware
//
#include <hip/hip_runtime.h>
#include <hip/hip_bf16.h>
#include <stdint.h>

// ---------------- problem constants ----------------
#define BB    64
#define CIN   128
#define COUT  128
#define TT_T  128
#define JJ    25
#define EPSBN 1e-5f

#define TCHUNK 8                    // t values per workgroup
#define NLOC   (TCHUNK * JJ)        // 200 flattened (t,j) columns
#define NPAD   208                  // 13 tiles of 16
#define NTILES (NPAD / 16)          // 13
#define UCOLS  (TCHUNK * 32)        // U stored with 32-col padding per t

typedef __attribute__((ext_vector_type(16))) __bf16          v16bf;
typedef __attribute__((ext_vector_type(16))) unsigned short  v16us;
typedef __attribute__((ext_vector_type(8)))  unsigned short  v8us;
typedef __attribute__((ext_vector_type(8)))  float           v8f;

union frag_u { v16us u; v16bf b; };

// round-to-nearest-even fp32 -> bf16 bits
__device__ __forceinline__ unsigned bf_rne(float f) {
    unsigned u = __float_as_uint(f);
    unsigned r = u + 0x7FFFu + ((u >> 16) & 1u);
    return r >> 16;
}
// split: hi = bf16(f), lo = bf16(f - hi)
__device__ __forceinline__ void split2(float f, unsigned short& hb, unsigned short& lb) {
    unsigned h = bf_rne(f);
    float    hf = __uint_as_float(h << 16);
    hb = (unsigned short)h;
    lb = (unsigned short)bf_rne(f - hf);
}
__device__ __forceinline__ unsigned pack_split(float f) {
    unsigned short h, l; split2(f, h, l);
    return ((unsigned)h << 16) | l;
}

__device__ __forceinline__ v8f wmma_bf16(v16us a, v16us b, v8f c) {
    frag_u A, B;
    A.u = a; B.u = b;
    return __builtin_amdgcn_wmma_f32_16x16x32_bf16(false, A.b, false, B.b,
                                                   (short)0, c, false, false);
}

__device__ __forceinline__ v16us cat8(v8us a, v8us b) {
    return __builtin_shufflevector(a, b, 0, 1, 2, 3, 4, 5, 6, 7,
                                         8, 9, 10, 11, 12, 13, 14, 15);
}

// ---------------- kernel 1: temporal mean  xt[b,c,j] = mean_t x[b,c,t,j] ----
__global__ __launch_bounds__(128)
void k_mean(const float* __restrict__ x, float* __restrict__ xt) {
    __shared__ float sj[JJ];
    const int bc = blockIdx.x;                  // b*CIN + c
    if (threadIdx.x < JJ) sj[threadIdx.x] = 0.f;
    __syncthreads();
    const float* p = x + (size_t)bc * (TT_T * JJ);
    for (int i = threadIdx.x; i < TT_T * JJ; i += blockDim.x)
        atomicAdd(&sj[i % JJ], p[i]);
    __syncthreads();
    if (threadIdx.x < JJ)
        xt[(size_t)bc * JJ + threadIdx.x] = sj[threadIdx.x] * (1.0f / TT_T);
}

// ---------------- kernel 2: attention -> per-batch M (packed bf16 split) ----
// M[b] = adj * (I + alpha*dyn[b]) padded to 32x32 ; cb[b,k] = 1 + alpha*colsum(dyn)
__global__ __launch_bounds__(256)
void k_attn(const float* __restrict__ xt, const float* __restrict__ adj,
            const float* __restrict__ Wk, const float* __restrict__ bk,
            const float* __restrict__ Wq, const float* __restrict__ bq,
            const float* __restrict__ alpha,
            unsigned* __restrict__ Mpk, float* __restrict__ cb) {
    __shared__ float xt_s[CIN * JJ];
    __shared__ float adj_s[JJ * JJ];
    __shared__ float kk_s[COUT * JJ];
    __shared__ float qq_s[COUT * JJ];
    __shared__ float dyn_s[JJ * JJ];
    const int b = blockIdx.x, tid = threadIdx.x;

    for (int i = tid; i < CIN * JJ; i += 256) xt_s[i] = xt[(size_t)b * CIN * JJ + i];
    for (int i = tid; i < JJ * JJ; i += 256)  adj_s[i] = adj[i];
    __syncthreads();

    for (int i = tid; i < COUT * JJ; i += 256) {
        const int o = i / JJ, j = i % JJ;
        float ak = bk[o], aq = bq[o];
        for (int c = 0; c < CIN; ++c) {
            const float xv = xt_s[c * JJ + j];
            ak = fmaf(Wk[o * CIN + c], xv, ak);
            aq = fmaf(Wq[o * CIN + c], xv, aq);
        }
        kk_s[i] = ak; qq_s[i] = aq;
    }
    __syncthreads();

    const float iscale = rsqrtf((float)COUT);
    for (int i = tid; i < JJ * JJ; i += 256) {
        const int jj = i / JJ, k = i % JJ;
        float s = 0.f;
        for (int o = 0; o < COUT; ++o)
            s = fmaf(qq_s[o * JJ + jj], kk_s[o * JJ + k], s);
        dyn_s[i] = s * iscale;
    }
    __syncthreads();

    if (tid < JJ) {                              // row-wise softmax
        float m = -1e30f;
        for (int k = 0; k < JJ; ++k) m = fmaxf(m, dyn_s[tid * JJ + k]);
        float sum = 0.f;
        for (int k = 0; k < JJ; ++k) {
            const float e = __expf(dyn_s[tid * JJ + k] - m);
            dyn_s[tid * JJ + k] = e; sum += e;
        }
        const float inv = 1.f / sum;
        for (int k = 0; k < JJ; ++k) dyn_s[tid * JJ + k] *= inv;
    }
    __syncthreads();

    const float al = alpha[0];
    for (int i = tid; i < 32 * 32; i += 256) {
        const int m = i >> 5, k = i & 31;
        float v = 0.f;
        if (m < JJ && k < JJ) {
            v = adj_s[m * JJ + k];
            for (int j = 0; j < JJ; ++j)
                v = fmaf(al * adj_s[m * JJ + j], dyn_s[j * JJ + k], v);
        }
        Mpk[((size_t)b * 32 + m) * 32 + k] = pack_split(v);
    }
    if (tid < 32) {
        float v = 0.f;
        if (tid < JJ) {
            v = 1.f;
            for (int j = 0; j < JJ; ++j) v = fmaf(al, dyn_s[j * JJ + tid], v);
        }
        cb[b * 32 + tid] = v;
    }
}

// ---------------- kernel 3: fused  out = relu(bn(Ws*x*M + bs*cb)) ----------
// LDS planes (all ushort bf16 bits, fragment-contiguous):
//   Xhi/Xlo[n][c]        n in [0,NPAD), c in [0,CIN)   (x transposed, split)
//   Uhi/Ulo[o][t*32+j]   o in [0,COUT), 32 cols per t  (U split)
__global__ __launch_bounds__(256)
void k_main(const float* __restrict__ x, const float* __restrict__ Ws,
            const float* __restrict__ bs, const unsigned* __restrict__ Mpk,
            const float* __restrict__ cb, const float* __restrict__ gamma,
            const float* __restrict__ beta, const float* __restrict__ rmean,
            const float* __restrict__ rvar, float* __restrict__ out) {
    extern __shared__ unsigned char smem[];
    unsigned short* Xhi = (unsigned short*)smem;            // NPAD*CIN
    unsigned short* Xlo = Xhi + NPAD * CIN;                 // NPAD*CIN
    unsigned short* Uhi = Xlo + NPAD * CIN;                 // COUT*UCOLS
    unsigned short* Ulo = Uhi + COUT * UCOLS;               // COUT*UCOLS
    float* scale_s = (float*)(Ulo + COUT * UCOLS);          // 128
    float* shift_s = scale_s + COUT;                        // 128
    float* bs_s    = shift_s + COUT;                        // 128
    float* cb_s    = bs_s + COUT;                           // 32

    const int b = blockIdx.y, t0 = blockIdx.x * TCHUNK;
    const int tid = threadIdx.x, lane = tid & 31, wave = tid >> 5;
    const int lane15 = lane & 15;
    const int hiHalf = (lane >= 16) ? 1 : 0;

    for (int i = tid; i < COUT; i += 256) {
        const float sc = gamma[i] * rsqrtf(rvar[i] + EPSBN);
        scale_s[i] = sc;
        shift_s[i] = beta[i] - rmean[i] * sc;
        bs_s[i]    = bs[i];
    }
    if (tid < 32) cb_s[tid] = cb[b * 32 + tid];

    // zero only padded U columns (j = 25..31 for each t) in both planes
    for (int i = tid; i < COUT * TCHUNK * 7; i += 256) {
        const int o = i / (TCHUNK * 7), r = i % (TCHUNK * 7);
        const int col = (r / 7) * 32 + 25 + (r % 7);
        Uhi[o * UCOLS + col] = 0;
        Ulo[o * UCOLS + col] = 0;
    }

    // load x tile [CIN][TCHUNK*JJ] -> split planes, transposed [n][c]
    for (int i = tid; i < CIN * NLOC; i += 256) {
        const int c = i / NLOC, n = i % NLOC;
        const float v = x[(((size_t)b * CIN + c) * TT_T + t0) * JJ + n];
        unsigned short h, l; split2(v, h, l);
        Xhi[n * CIN + c] = h;
        Xlo[n * CIN + c] = l;
    }

    // A fragments of Ws (o-tile = wave), per documented 16-bit 16x32 A layout
    v16us wsA_hi[4], wsA_lo[4];
    {
        const int orow = wave * 16 + lane15;
        const int koff = hiHalf ? 8 : 0;
        for (int kt = 0; kt < 4; ++kt) {
            for (int h = 0; h < 16; ++h) {
                const int kk2 = ((h < 8) ? 0 : 16) + (((h >> 1) & 3) * 2) + (h & 1) + koff;
                unsigned short hb, lb;
                split2(Ws[orow * CIN + kt * 32 + kk2], hb, lb);
                wsA_hi[kt][h] = hb;
                wsA_lo[kt][h] = lb;
            }
        }
    }
    // B fragments of M[b] (32x16 per k-out tile), per 16-bit B layout
    v16us mB_hi[2], mB_lo[2];
    for (int kt2 = 0; kt2 < 2; ++kt2) {
        for (int h = 0; h < 16; ++h) {
            const int m = ((h < 8) ? 0 : 16) + (h & 7) + (hiHalf ? 8 : 0);
            const unsigned pk = Mpk[((size_t)b * 32 + m) * 32 + kt2 * 16 + lane15];
            mB_hi[kt2][h] = (unsigned short)(pk >> 16);
            mB_lo[kt2][h] = (unsigned short)(pk & 0xFFFFu);
        }
    }
    __syncthreads();

    // -------- stage 1: U = Ws * x  (K = CIN = 128, split-bf16 3-term) --------
    const int off16 = hiHalf ? 8 : 0;
    for (int nt = 0; nt < NTILES; ++nt) {
        v8f acc = {};
        const int ncol = nt * 16 + lane15;
        const unsigned short* rowH = &Xhi[ncol * CIN + off16];
        const unsigned short* rowL = &Xlo[ncol * CIN + off16];
        for (int kt = 0; kt < 4; ++kt) {
            const v16us bhi = cat8(*(const v8us*)(rowH + kt * 32), *(const v8us*)(rowH + kt * 32 + 16));   // two 16B chunks
            const v16us blo = cat8(*(const v8us*)(rowL + kt * 32), *(const v8us*)(rowL + kt * 32 + 16));
            acc = wmma_bf16(wsA_hi[kt], bhi, acc);
            acc = wmma_bf16(wsA_hi[kt], blo, acc);
            acc = wmma_bf16(wsA_lo[kt], bhi, acc);
        }
        if (ncol < NLOC) {                       // write U split planes
            const int t = ncol / JJ, j = ncol % JJ, col = t * 32 + j;
            const int obase = wave * 16 + (hiHalf ? 8 : 0);
            for (int v = 0; v < 8; ++v) {
                unsigned short hb, lb; split2(acc[v], hb, lb);
                Uhi[(obase + v) * UCOLS + col] = hb;
                Ulo[(obase + v) * UCOLS + col] = lb;
            }
        }
    }
    __syncthreads();

    // -------- stage 2: out = U * M + bs*cb ; BN ; ReLU  (per t, K = 25->32) --
    const int orow = wave * 16 + lane15;
    const int koff = hiHalf ? 8 : 0;
    for (int t = 0; t < TCHUNK; ++t) {
        const unsigned short* ph = &Uhi[orow * UCOLS + t * 32 + koff];
        const unsigned short* pl = &Ulo[orow * UCOLS + t * 32 + koff];
        const v16us ahi = cat8(*(const v8us*)ph, *(const v8us*)(ph + 16));
        const v16us alo = cat8(*(const v8us*)pl, *(const v8us*)(pl + 16));
        for (int kt2 = 0; kt2 < 2; ++kt2) {
            v8f acc = {};
            acc = wmma_bf16(ahi, mB_hi[kt2], acc);
            acc = wmma_bf16(ahi, mB_lo[kt2], acc);
            acc = wmma_bf16(alo, mB_hi[kt2], acc);
            const int kcol = kt2 * 16 + lane15;
            if (kcol < JJ) {
                const int ob = wave * 16 + (hiHalf ? 8 : 0);
                const float cbv = cb_s[kcol];
                for (int v = 0; v < 8; ++v) {
                    const int o = ob + v;
                    float val = acc[v] + bs_s[o] * cbv;
                    val = fmaf(val, scale_s[o], shift_s[o]);
                    out[(((size_t)b * COUT + o) * TT_T + (t0 + t)) * JJ + kcol] =
                        fmaxf(val, 0.f);
                }
            }
        }
    }
}

// ---------------- host launch ----------------
extern "C" void kernel_launch(void* const* d_in, const int* in_sizes, int n_in,
                              void* d_out, int out_size, void* d_ws, size_t ws_size,
                              hipStream_t stream) {
    (void)in_sizes; (void)n_in; (void)out_size; (void)ws_size;
    const float* x     = (const float*)d_in[0];
    const float* adj   = (const float*)d_in[1];
    const float* Wk    = (const float*)d_in[2];
    const float* bk    = (const float*)d_in[3];
    const float* Wq    = (const float*)d_in[4];
    const float* bq    = (const float*)d_in[5];
    const float* Ws    = (const float*)d_in[6];
    const float* bs    = (const float*)d_in[7];
    const float* gamma = (const float*)d_in[8];
    const float* beta  = (const float*)d_in[9];
    const float* rmean = (const float*)d_in[10];
    const float* rvar  = (const float*)d_in[11];
    const float* alpha = (const float*)d_in[12];
    float* out = (float*)d_out;

    char* ws = (char*)d_ws;
    float*    xt  = (float*)ws;                                   // B*CIN*J
    unsigned* Mpk = (unsigned*)(ws + (size_t)BB * CIN * JJ * 4);  // B*32*32
    float*    cbp = (float*)((char*)Mpk + (size_t)BB * 32 * 32 * 4); // B*32

    k_mean<<<BB * CIN, 128, 0, stream>>>(x, xt);
    k_attn<<<BB, 256, 0, stream>>>(xt, adj, Wk, bk, Wq, bq, alpha, Mpk, cbp);

    const size_t lds = (size_t)(2 * NPAD * CIN + 2 * COUT * UCOLS) * 2
                     + (size_t)(3 * COUT + 32) * 4;
    dim3 grid(TT_T / TCHUNK, BB);
    k_main<<<grid, 256, lds, stream>>>(x, Ws, bs, Mpk, cbp, gamma, beta,
                                       rmean, rvar, out);
}
